// SNNMaxPool2d_1692217115110
// MI455X (gfx1250) — matrix operand order repaired and checked by reading, not compiled, not run on hardware
//
#include <hip/hip_runtime.h>

typedef __attribute__((ext_vector_type(16))) _Float16 v16h;
typedef __attribute__((ext_vector_type(8)))  _Float16 v8h;
typedef __attribute__((ext_vector_type(8)))  float    v8f;

#define TT   100        // timesteps
#define ROWP 136        // LDS row pitch in halfs (128 padded T + 8 anti-bank-conflict)
#define WIN_PER_WAVE 16
#define WAVES_PER_BLOCK 8

// x: [B=4, C=32, H=64, W=64, T=100] f32 spikes (0/1)
// out: [4, 32, 32, 32, 100] f32 one-hot of earliest spike per 2x2 window
__global__ __launch_bounds__(256) void snn_maxpool_wta_kernel(
    const float* __restrict__ x, float* __restrict__ out)
{
    __shared__ _Float16 smem[WAVES_PER_BLOCK * WIN_PER_WAVE * ROWP];

    const int lane  = threadIdx.x & 31;
    const int wave  = threadIdx.x >> 5;
    const int half_ = lane >> 4;   // which 16-lane half
    const int sub   = lane & 15;   // M row (A) and N column (B/C/D)

    _Float16* srow = &smem[wave * WIN_PER_WAVE * ROWP];

    // 16 consecutive flattened windows per wave; w = ((b*32+c)*32+oh)*32+ow
    const int w0   = blockIdx.x * (WAVES_PER_BLOCK * WIN_PER_WAVE) + wave * WIN_PER_WAVE;
    const int ow0  = w0 & 31;
    const int rest = w0 >> 5;
    const int oh   = rest & 31;
    const int bc   = rest >> 5;                           // b*32 + c
    const long rb0 = ((long)bc * 64 + 2 * oh) * 64 * TT;  // input row 2*oh
    const long rb1 = rb0 + (long)64 * TT;                 // input row 2*oh+1

    // ---- Phase 1: 2x2 window sums -> f16 in LDS, T padded to 128 with zeros ----
    for (int i = 0; i < WIN_PER_WAVE; ++i) {
        const long col  = (long)(2 * (ow0 + i)) * TT;
        const float* p00 = x + rb0 + col;
        const float* p01 = p00 + TT;
        const float* p10 = x + rb1 + col;
        const float* p11 = p10 + TT;
        _Float16* dst = srow + i * ROWP;
#pragma unroll
        for (int j = 0; j < 4; ++j) {
            const int t = lane + 32 * j;                  // coalesced 128B per load
            float s = 0.0f;
            if (t < TT)                                   // streamed once: NT loads
                s = __builtin_nontemporal_load(p00 + t)
                  + __builtin_nontemporal_load(p01 + t)
                  + __builtin_nontemporal_load(p10 + t)
                  + __builtin_nontemporal_load(p11 + t);
            dst[t] = (_Float16)s;                         // zero padding for t in [100,128)
        }
    }

    __syncthreads();

    // ---- Phase 2: hoist A fragments (16 windows x K=128, four 16x32 f16 tiles) ----
    // A k order used here (contiguous, same as B): lane<16 holds M=sub, K in {0..15}; lane>=16 same M, K in {16..31}
    v16h A[4];
#pragma unroll
    for (int kc = 0; kc < 4; ++kc) {
        const _Float16* base = srow + sub * ROWP + 32 * kc + 16 * half_;
        const v8h lo = *(const v8h*)(base);               // ds_load_b128, conflict-free
        const v8h hi = *(const v8h*)(base + 8);
        A[kc] = __builtin_shufflevector(lo, hi, 0, 1, 2, 3, 4, 5, 6, 7,
                                                 8, 9, 10, 11, 12, 13, 14, 15);
    }

    // ---- Precompute the only 5 distinct triangular B fragments ----
    // B layout: lane's N = sub; element e holds local K = 16*half_ + e.
    // Block offset d = 16*nc - 32*kc: d>=32 -> all-ones; d in {0,16} -> partial.
    v16h B_one, Bi0, Bs0, Bi16, Bs16;
#pragma unroll
    for (int e = 0; e < 16; ++e) {
        const int tp = 16 * half_ + e;                    // local t' row index
        B_one[e] = (_Float16)1.0f;
        Bi0[e]   = (tp <= sub)      ? (_Float16)1.0f : (_Float16)0.0f;  // d=0, t'<=t
        Bs0[e]   = (tp <  sub)      ? (_Float16)1.0f : (_Float16)0.0f;  // d=0, t'< t
        Bi16[e]  = (tp <= 16 + sub) ? (_Float16)1.0f : (_Float16)0.0f;  // d=16, t'<=t
        Bs16[e]  = (tp <  16 + sub) ? (_Float16)1.0f : (_Float16)0.0f;  // d=16, t'< t
    }

    // ---- Phase 3: inclusive & strict cumsum tiles via WMMA, emit one-hot ----
    v8f c_ones = {0.f, 0.f, 0.f, 0.f, 0.f, 0.f, 0.f, 0.f}; // sum over all-ones K-chunks
#pragma unroll
    for (int nc = 0; nc < 7; ++nc) {                      // 7 N-tiles cover t = 0..111
        if ((nc & 1) == 0 && nc >= 2)                     // chunk (nc/2 - 1) becomes all-ones
            c_ones = __builtin_amdgcn_wmma_f32_16x16x32_f16(
                         false, A[(nc >> 1) - 1], false, B_one, (short)0, c_ones,
                         false, false);
        const int kp = nc >> 1;                           // partial (diagonal) K-chunk
        v8f ci, cs;
        if (nc & 1) {
            ci = __builtin_amdgcn_wmma_f32_16x16x32_f16(
                     false, A[kp], false, Bi16, (short)0, c_ones, false, false);
            cs = __builtin_amdgcn_wmma_f32_16x16x32_f16(
                     false, A[kp], false, Bs16, (short)0, c_ones, false, false);
        } else {
            ci = __builtin_amdgcn_wmma_f32_16x16x32_f16(
                     false, A[kp], false, Bi0, (short)0, c_ones, false, false);
            cs = __builtin_amdgcn_wmma_f32_16x16x32_f16(
                     false, A[kp], false, Bs0, (short)0, c_ones, false, false);
        }
        const int tcol = 16 * nc + sub;                   // this lane's global t
        if (tcol < TT) {
#pragma unroll
            for (int r = 0; r < 8; ++r) {                 // C layout: VGPR r -> M = r + 8*half_
                const int M = r + 8 * half_;
                // one-hot: cumsum positive AND strict cumsum still zero
                const float o = (ci[r] > 0.0f && cs[r] <= 0.0f) ? 1.0f : 0.0f;
                __builtin_nontemporal_store(o, out + (long)(w0 + M) * TT + tcol);
            }
        }
    }
}

extern "C" void kernel_launch(void* const* d_in, const int* in_sizes, int n_in,
                              void* d_out, int out_size, void* d_ws, size_t ws_size,
                              hipStream_t stream) {
    const float* x  = (const float*)d_in[0];
    float* out      = (float*)d_out;
    const int num_windows = 4 * 32 * 32 * 32;                          // 131072
    const int blocks = num_windows / (WAVES_PER_BLOCK * WIN_PER_WAVE); // 1024
    snn_maxpool_wta_kernel<<<blocks, 256, 0, stream>>>(x, out);
}
